// Preprocessor_84129819394499
// MI455X (gfx1250) — matrix operand order repaired and checked by reading, not compiled, not run on hardware
//
#include <hip/hip_runtime.h>
#include <hip/hip_bf16.h>

// ---------------------------------------------------------------------------
// GRU preprocessor for gfx1250 (MI455X), bf16 WMMA + f32 gates.
//   seq_out [256,2048] followed by cmd_out [256,16,1024] in d_out (f32).
// 32-row M-tiles: each B fragment (weight tile, streamed from L2) feeds two
// WMMAs -> halves per-timestep L2 weight traffic on the sequential recurrence.
// ---------------------------------------------------------------------------

typedef __attribute__((ext_vector_type(16))) __bf16 v16bf;
typedef __attribute__((ext_vector_type(8)))  float  v8f;

union Frag16 { v16bf v; uint4 q[2]; };

#define HID3    3072
#define EMBW    512
#define HIDW    1024
#define KW      1536          // EMBW + HIDW (fused [x_t | h] K dimension)
#define KPAD    1552          // padded LDS row to dodge bank conflicts
#define ROWS    32            // batch rows per block (two WMMA M tiles)
#define SEQ_OUT_ELEMS (256 * 2048)

// ------------------------- one-time bf16 casts ------------------------------

__global__ void cast_bf16_kernel(const float* __restrict__ src,
                                 __bf16* __restrict__ dst, int n) {
    int i = blockIdx.x * blockDim.x + threadIdx.x;
    if (i < n) dst[i] = (__bf16)src[i];
}

// Build concatenated weight Wc[n][k]: k<512 -> Wih[n][k], else Whh[n][k-512]
__global__ void build_wc_kernel(const float* __restrict__ Wih,
                                const float* __restrict__ Whh,
                                __bf16* __restrict__ wc) {
    int idx = blockIdx.x * blockDim.x + threadIdx.x;
    if (idx >= HID3 * KW) return;
    int n = idx / KW, k = idx % KW;
    float v = (k < EMBW) ? Wih[n * EMBW + k] : Whh[n * HIDW + (k - EMBW)];
    wc[idx] = (__bf16)v;
}

// -------------------- fragment loaders (wave32 layouts) ---------------------

// A 16x32 bf16 tile from LDS: lane(m = lane&15, half = lane>>4);
// VGPR0-3 hold K = 8*half + 0..7, VGPR4-7 hold K = 16 + 8*half + 0..7.
static __device__ __forceinline__
v16bf load_a_frag(const __bf16* As_row, int k0, int lh) {
    Frag16 a;
    const __bf16* ap = As_row + k0 + 8 * lh;
    a.q[0] = *(const uint4*)ap;
    a.q[1] = *(const uint4*)(ap + 16);
    return a.v;
}

// B 32x16 bf16 tile from row-major W[n][k]: lane(n = n0 + lane&15, half);
// VGPR0-3 hold K = 8*half + 0..7, VGPR4-7 hold K = 16 + 8*half + 0..7.
static __device__ __forceinline__
v16bf load_b_frag(const __bf16* __restrict__ Wc, int nrow, int k0, int lm, int lh) {
    Frag16 b;
    const __bf16* wp = Wc + (size_t)(nrow + lm) * KW + k0 + 8 * lh;
    b.q[0] = *(const uint4*)wp;
    b.q[1] = *(const uint4*)(wp + 16);
    return b.v;
}

#define WMMA_BF16(A, B, C) \
    __builtin_amdgcn_wmma_f32_16x16x32_bf16(false, (A), false, (B), (short)0, (C), false, false)

// ------------------------------ main GRU ------------------------------------
// Blocks 0..15  : encoders (blk>>3 = enc id, 32 batch rows each)
// Blocks 16..143: command GRU (32 of 4096 flattened sequences each)

__global__ __launch_bounds__(512, 1)
void gru_fused_kernel(const __bf16* __restrict__ embb,
                      const __bf16* __restrict__ wc_e0,
                      const __bf16* __restrict__ wc_e1,
                      const __bf16* __restrict__ wc_cmd,
                      const int* __restrict__ obs_tokens,
                      const int* __restrict__ obs_lens,
                      const int* __restrict__ cmd_tokens,
                      const int* __restrict__ cmd_lens,
                      const float* __restrict__ bih_enc,
                      const float* __restrict__ bhh_enc,
                      const float* __restrict__ bih_cmd,
                      const float* __restrict__ bhh_cmd,
                      float* __restrict__ out) {
    __shared__ __bf16 As[ROWS][KPAD];   // [x_t | h] bf16 A tile (32 rows)
    __shared__ int    lens_sh[ROWS];

    const int tid  = threadIdx.x;
    const int wave = tid >> 5;    // 0..15, each owns 64 hidden columns
    const int lane = tid & 31;
    const int lm   = lane & 15;   // M (A) / N (B,C) index within tile
    const int lh   = lane >> 4;   // lane half

    // -------- per-block role --------
    const __bf16* Wc;  const int* toks; const int* lens;
    const float *bih, *bhh;  float* outp;
    int L, tok_stride, out_stride;
    const int blk = blockIdx.x;
    if (blk < 16) {
        const int enc  = blk >> 3;
        const int row0 = (blk & 7) * ROWS;
        Wc   = enc ? wc_e1 : wc_e0;
        toks = obs_tokens + (size_t)enc * 256 * 128 + (size_t)row0 * 128;
        lens = obs_lens + enc * 256 + row0;
        bih  = bih_enc + enc * HID3;
        bhh  = bhh_enc + enc * HID3;
        L = 128; tok_stride = 128;
        outp = out + (size_t)row0 * 2048 + enc * 1024;
        out_stride = 2048;
    } else {
        const int q0 = (blk - 16) * ROWS;     // flattened (b, cmd) row
        Wc   = wc_cmd;
        toks = cmd_tokens + (size_t)q0 * 16;
        lens = cmd_lens + q0;
        bih  = bih_cmd; bhh = bhh_cmd;
        L = 16; tok_stride = 16;
        outp = out + SEQ_OUT_ELEMS + (size_t)q0 * 1024;
        out_stride = 1024;
    }

    if (tid < ROWS) lens_sh[tid] = lens[tid];

    // h in registers: hreg[s][mt][v] = h[m = mt*16 + v + 8*lh][j = wave*64 + s*16 + lm]
    float hreg[4][2][8];
    #pragma unroll
    for (int s = 0; s < 4; ++s)
        #pragma unroll
        for (int mt = 0; mt < 2; ++mt)
            #pragma unroll
            for (int v = 0; v < 8; ++v) hreg[s][mt][v] = 0.f;
    __syncthreads();

    for (int t = 0; t < L; ++t) {
        // ---- stage 1: build A = [emb[token_t] | bf16(h)] in LDS ----
        {
            const int r = tid >> 4, c = tid & 15;   // 32 rows x 16 chunks
            const int tok = toks[r * tok_stride + t];
            const uint4* src = (const uint4*)(embb + (size_t)tok * EMBW + c * 32);
            uint4* dst = (uint4*)&As[r][c * 32];
            dst[0] = src[0]; dst[1] = src[1]; dst[2] = src[2]; dst[3] = src[3];
        }
        #pragma unroll
        for (int s = 0; s < 4; ++s) {
            const int j = wave * 64 + s * 16 + lm;
            #pragma unroll
            for (int mt = 0; mt < 2; ++mt)
                #pragma unroll
                for (int v = 0; v < 8; ++v)
                    As[mt * 16 + v + 8 * lh][EMBW + j] = (__bf16)hreg[s][mt][v];
        }
        __syncthreads();

        // ---- stage 2: gates = A @ Wc^T; each B fragment feeds two M tiles ----
        for (int s = 0; s < 4; ++s) {
            const int j0 = wave * 64 + s * 16;
            v8f ar0 = {}, az0 = {}, ain0 = {}, ahn0 = {};
            v8f ar1 = {}, az1 = {}, ain1 = {}, ahn1 = {};

            // K in [0, 512): input projection part; n-gate accumulates i_n.
            #pragma unroll 2
            for (int k0 = 0; k0 < EMBW; k0 += 32) {
                const v16bf a0 = load_a_frag(&As[lm][0],      k0, lh);
                const v16bf a1 = load_a_frag(&As[16 + lm][0], k0, lh);
                const v16bf br = load_b_frag(Wc, j0,            k0, lm, lh);
                const v16bf bz = load_b_frag(Wc, HIDW + j0,     k0, lm, lh);
                const v16bf bn = load_b_frag(Wc, 2 * HIDW + j0, k0, lm, lh);
                ar0  = WMMA_BF16(a0, br, ar0);   ar1  = WMMA_BF16(a1, br, ar1);
                az0  = WMMA_BF16(a0, bz, az0);   az1  = WMMA_BF16(a1, bz, az1);
                ain0 = WMMA_BF16(a0, bn, ain0);  ain1 = WMMA_BF16(a1, bn, ain1);
            }
            // K in [512, 1536): hidden projection part; n-gate accumulates h_n.
            #pragma unroll 2
            for (int k0 = EMBW; k0 < KW; k0 += 32) {
                const v16bf a0 = load_a_frag(&As[lm][0],      k0, lh);
                const v16bf a1 = load_a_frag(&As[16 + lm][0], k0, lh);
                const v16bf br = load_b_frag(Wc, j0,            k0, lm, lh);
                const v16bf bz = load_b_frag(Wc, HIDW + j0,     k0, lm, lh);
                const v16bf bn = load_b_frag(Wc, 2 * HIDW + j0, k0, lm, lh);
                ar0  = WMMA_BF16(a0, br, ar0);   ar1  = WMMA_BF16(a1, br, ar1);
                az0  = WMMA_BF16(a0, bz, az0);   az1  = WMMA_BF16(a1, bz, az1);
                ahn0 = WMMA_BF16(a0, bn, ahn0);  ahn1 = WMMA_BF16(a1, bn, ahn1);
            }

            const int j = j0 + lm;
            const float br_i = bih[j],            br_h = bhh[j];
            const float bz_i = bih[HIDW + j],     bz_h = bhh[HIDW + j];
            const float bn_i = bih[2 * HIDW + j], bn_h = bhh[2 * HIDW + j];
            #pragma unroll
            for (int mt = 0; mt < 2; ++mt) {
                const v8f& ar  = mt ? ar1 : ar0;
                const v8f& az  = mt ? az1 : az0;
                const v8f& ain = mt ? ain1 : ain0;
                const v8f& ahn = mt ? ahn1 : ahn0;
                #pragma unroll
                for (int v = 0; v < 8; ++v) {
                    const int mr = mt * 16 + v + 8 * lh;
                    const float hold = hreg[s][mt][v];
                    const float rg = 1.f / (1.f + __expf(-(ar[v] + br_i + br_h)));
                    const float zg = 1.f / (1.f + __expf(-(az[v] + bz_i + bz_h)));
                    const float ng = tanhf(ain[v] + bn_i + rg * (ahn[v] + bn_h));
                    const float hn = (1.f - zg) * ng + zg * hold;
                    hreg[s][mt][v] = (t < lens_sh[mr]) ? hn : hold;
                }
            }
        }
        __syncthreads();
    }

    // ---- write final hidden state ----
    #pragma unroll
    for (int s = 0; s < 4; ++s) {
        const int j = wave * 64 + s * 16 + lm;
        #pragma unroll
        for (int mt = 0; mt < 2; ++mt)
            #pragma unroll
            for (int v = 0; v < 8; ++v) {
                const int m = mt * 16 + v + 8 * lh;
                outp[(size_t)m * out_stride + j] = hreg[s][mt][v];
            }
    }
}

// ------------------------------- launch -------------------------------------

extern "C" void kernel_launch(void* const* d_in, const int* in_sizes, int n_in,
                              void* d_out, int out_size, void* d_ws, size_t ws_size,
                              hipStream_t stream) {
    const int*   obs_tokens = (const int*)d_in[0];
    const int*   obs_lens   = (const int*)d_in[1];
    const int*   cmd_tokens = (const int*)d_in[2];
    const int*   cmd_lens   = (const int*)d_in[3];
    const float* emb        = (const float*)d_in[4];
    const float* Wih_enc    = (const float*)d_in[5];
    const float* Whh_enc    = (const float*)d_in[6];
    const float* bih_enc    = (const float*)d_in[7];
    const float* bhh_enc    = (const float*)d_in[8];
    const float* Wih_cmd    = (const float*)d_in[9];
    const float* Whh_cmd    = (const float*)d_in[10];
    const float* bih_cmd    = (const float*)d_in[11];
    const float* bhh_cmd    = (const float*)d_in[12];
    float* out = (float*)d_out;

    // workspace layout (all 256B-aligned offsets)
    const size_t EMB_BYTES = (size_t)20000 * EMBW * sizeof(__bf16);   // 20,480,000
    const size_t WC_BYTES  = (size_t)HID3 * KW * sizeof(__bf16);      //  9,437,184
    char* ws = (char*)d_ws;
    __bf16* embb   = (__bf16*)ws;
    __bf16* wc_e0  = (__bf16*)(ws + EMB_BYTES);
    __bf16* wc_e1  = (__bf16*)(ws + EMB_BYTES + WC_BYTES);
    __bf16* wc_cmd = (__bf16*)(ws + EMB_BYTES + 2 * WC_BYTES);

    // 1) cast embedding table to bf16
    {
        const int n = 20000 * EMBW;
        cast_bf16_kernel<<<(n + 255) / 256, 256, 0, stream>>>(emb, embb, n);
    }
    // 2) build fused [Wih | Whh] bf16 weights for enc0, enc1, cmd
    {
        const int n = HID3 * KW;
        const int g = (n + 255) / 256;
        build_wc_kernel<<<g, 256, 0, stream>>>(Wih_enc, Whh_enc, wc_e0);
        build_wc_kernel<<<g, 256, 0, stream>>>(Wih_enc + (size_t)HID3 * EMBW,
                                               Whh_enc + (size_t)HID3 * HIDW, wc_e1);
        build_wc_kernel<<<g, 256, 0, stream>>>(Wih_cmd, Whh_cmd, wc_cmd);
    }
    // 3) fused GRU recurrence: 16 encoder blocks + 128 command blocks
    gru_fused_kernel<<<144, 512, 0, stream>>>(embb, wc_e0, wc_e1, wc_cmd,
                                              obs_tokens, obs_lens, cmd_tokens, cmd_lens,
                                              bih_enc, bhh_enc, bih_cmd, bhh_cmd, out);
}
